// GNN_40424232190376
// MI455X (gfx1250) — matrix operand order repaired and checked by reading, not compiled, not run on hardware
//
#include <hip/hip_runtime.h>

#define NN 100000   // nodes
#define NE 1600000  // edges
#define D  128      // feature dim
#define NL 3        // GIN layers
#define NG 512      // graphs
#define NC 10       // classes

typedef __attribute__((ext_vector_type(16))) __bf16 v16bf;
typedef __attribute__((ext_vector_type(8)))  float  v8f;

__device__ __forceinline__ unsigned short f2bf(float f) {
  unsigned int u = __float_as_uint(f);
  u += 0x7fffu + ((u >> 16) & 1u);           // round-to-nearest-even
  return (unsigned short)(u >> 16);
}

union Frag {
  uint4 q[2];
  v16bf v;
};

// ---------------- f32 -> bf16 weight conversion ----------------
__global__ void cvt_bf16_kernel(const float* __restrict__ in,
                                unsigned short* __restrict__ out, int n) {
  int i = blockIdx.x * blockDim.x + threadIdx.x;
  if (i < n) out[i] = f2bf(in[i]);
}

// ---------------- edge scatter-add: aggr[dst] += h[src] ----------------
// one lane handles 4 dims; 32 lanes cover the 128-dim row of one edge
__global__ void scatter_kernel(const float* __restrict__ h,
                               const int* __restrict__ src,
                               const int* __restrict__ dst,
                               float* __restrict__ aggr) {
  int idx = blockIdx.x * 256 + threadIdx.x;
  int e = idx >> 5;
  if (e >= NE) return;
  int c = (idx & 31) * 4;
  int s = src[e], d = dst[e];
  float4 v = *(const float4*)&h[s * D + c];
  float* a = &aggr[d * D + c];
  unsafeAtomicAdd(a + 0, v.x);
  unsafeAtomicAdd(a + 1, v.y);
  unsafeAtomicAdd(a + 2, v.z);
  unsafeAtomicAdd(a + 3, v.w);
}

// ---------------- fused GIN MLP tile kernel ----------------
// out = (relu?)( relu( (hin+aggr) @ W1 + b1 ) @ W2 + b2 )
// 128 rows per block, 16 rows per wave, bf16 WMMA with f32 accumulate.
__global__ __launch_bounds__(256) void mlp_kernel(
    const float* __restrict__ hin, const float* __restrict__ aggr,
    const unsigned short* __restrict__ W1, const float* __restrict__ b1,
    const unsigned short* __restrict__ W2, const float* __restrict__ b2,
    float* __restrict__ hout, int nRows, int finalRelu) {
  constexpr int SZ = 136;  // padded bf16 row stride: 16B-aligned rows
  __shared__ __attribute__((aligned(16))) unsigned short zs[128 * SZ];

  const int tid    = threadIdx.x;
  const int lane   = tid & 31;
  const int wave   = tid >> 5;
  const int laneHi = lane >> 4;     // 0: lanes 0-15, 1: lanes 16-31
  const int laneLo = lane & 15;
  const int m0     = blockIdx.x * 128;

  // phase 0: z = hin (+ aggr), convert to bf16 into LDS (coalesced float4)
  for (int i = tid; i < 128 * 32; i += 256) {
    int r = i >> 5;
    int c = (i & 31) * 4;
    int row = m0 + r;
    float4 v = make_float4(0.f, 0.f, 0.f, 0.f);
    if (row < nRows) {
      v = *(const float4*)&hin[row * D + c];
      if (aggr) {
        float4 a = *(const float4*)&aggr[row * D + c];
        v.x += a.x; v.y += a.y; v.z += a.z; v.w += a.w;
      }
    }
    unsigned short* p = &zs[r * SZ + c];
    p[0] = f2bf(v.x); p[1] = f2bf(v.y); p[2] = f2bf(v.z); p[3] = f2bf(v.w);
  }
  __syncthreads();

  const int mw   = wave * 16;            // wave-private 16-row block in tile
  const int arow = mw + laneLo;          // A-frag LDS row for this lane

  // ---- GEMM1: hidden = relu(z @ W1 + b1) -> back into zs (own rows) ----
  Frag fa[4];
#pragma unroll
  for (int ks = 0; ks < 4; ++ks) {       // ISA 16-bit A 16x32 layout
    int base = arow * SZ + ks * 32 + 8 * laneHi;
    fa[ks].q[0] = *(const uint4*)&zs[base];        // K +0..7  (hi: +8..15)
    fa[ks].q[1] = *(const uint4*)&zs[base + 16];   // K +16..23 (hi: +24..31)
  }
#pragma unroll
  for (int nt = 0; nt < 8; ++nt) {
    const int n0 = nt * 16;
    v8f acc = {0.f, 0.f, 0.f, 0.f, 0.f, 0.f, 0.f, 0.f};
#pragma unroll
    for (int ks = 0; ks < 4; ++ks) {
      Frag fragB;                        // lane = N col, 16 K values (table order)
#pragma unroll
      for (int s = 0; s < 16; ++s) {
        int k = ks * 32 + 8 * laneHi + (s < 8 ? s : 8 + s);
        fragB.v[s] = __builtin_bit_cast(__bf16, W1[k * D + n0 + laneLo]);
      }
      acc = __builtin_amdgcn_wmma_f32_16x16x32_bf16(
          false, fa[ks].v, false, fragB.v, (short)0, acc, false, false);
    }
    const int col = n0 + laneLo;
    const float bias = b1[col];
#pragma unroll
    for (int r = 0; r < 8; ++r) {        // C layout: vgpr r -> row r (+8 hi)
      float v = acc[r] + bias;
      v = v > 0.f ? v : 0.f;
      zs[(mw + r + 8 * laneHi) * SZ + col] = f2bf(v);
    }
  }

  // ---- GEMM2: out = hidden @ W2 + b2 (optional relu) ----
#pragma unroll
  for (int ks = 0; ks < 4; ++ks) {
    int base = arow * SZ + ks * 32 + 8 * laneHi;
    fa[ks].q[0] = *(const uint4*)&zs[base];
    fa[ks].q[1] = *(const uint4*)&zs[base + 16];
  }
#pragma unroll
  for (int nt = 0; nt < 8; ++nt) {
    const int n0 = nt * 16;
    v8f acc = {0.f, 0.f, 0.f, 0.f, 0.f, 0.f, 0.f, 0.f};
#pragma unroll
    for (int ks = 0; ks < 4; ++ks) {
      Frag fragB;
#pragma unroll
      for (int s = 0; s < 16; ++s) {
        int k = ks * 32 + 8 * laneHi + (s < 8 ? s : 8 + s);
        fragB.v[s] = __builtin_bit_cast(__bf16, W2[k * D + n0 + laneLo]);
      }
      acc = __builtin_amdgcn_wmma_f32_16x16x32_bf16(
          false, fa[ks].v, false, fragB.v, (short)0, acc, false, false);
    }
    const int col = n0 + laneLo;
    const float bias = b2[col];
#pragma unroll
    for (int r = 0; r < 8; ++r) {
      int row = m0 + mw + r + 8 * laneHi;
      float v = acc[r] + bias;
      if (finalRelu) v = v > 0.f ? v : 0.f;
      if (row < nRows) hout[row * D + col] = v;
    }
  }
}

// ---------------- global add pool: xpool[batch[n]] += h[n] ----------------
__global__ void pool_kernel(const float* __restrict__ h,
                            const int* __restrict__ batch,
                            float* __restrict__ xpool) {
  int idx = blockIdx.x * 256 + threadIdx.x;
  int n = idx >> 5;
  if (n >= NN) return;
  int c = (idx & 31) * 4;
  int g = batch[n];
  float4 v = *(const float4*)&h[n * D + c];
  float* p = &xpool[g * D + c];
  unsafeAtomicAdd(p + 0, v.x);
  unsafeAtomicAdd(p + 1, v.y);
  unsafeAtomicAdd(p + 2, v.z);
  unsafeAtomicAdd(p + 3, v.w);
}

// ---------------- final fc: p_out = xpool @ fw + fb (128x10, scalar) -----
__global__ void fc_kernel(const float* __restrict__ xpool,
                          const float* __restrict__ fw,
                          const float* __restrict__ fbias,
                          float* __restrict__ pout) {
  int i = blockIdx.x * blockDim.x + threadIdx.x;
  if (i >= NG * NC) return;
  int g = i / NC, c = i % NC;
  float acc = fbias[c];
#pragma unroll 8
  for (int k = 0; k < D; ++k) acc += xpool[g * D + k] * fw[k * NC + c];
  pout[i] = acc;
}

extern "C" void kernel_launch(void* const* d_in, const int* in_sizes, int n_in,
                              void* d_out, int out_size, void* d_ws, size_t ws_size,
                              hipStream_t stream) {
  const float* x     = (const float*)d_in[0];
  const int*   ei    = (const int*)d_in[1];
  const int*   src   = ei;
  const int*   dst   = ei + NE;
  const int*   batch = (const int*)d_in[2];
  const float* w1    = (const float*)d_in[3];
  const float* b1    = (const float*)d_in[4];
  const float* w2    = (const float*)d_in[5];
  const float* b2    = (const float*)d_in[6];
  const float* pw1   = (const float*)d_in[7];
  const float* pb1   = (const float*)d_in[8];
  const float* pw2   = (const float*)d_in[9];
  const float* pb2   = (const float*)d_in[10];
  const float* fw    = (const float*)d_in[11];
  const float* fbias = (const float*)d_in[12];
  float* out = (float*)d_out;

  // workspace layout (all 16B-aligned offsets)
  char* ws = (char*)d_ws;
  float* aggr  = (float*)ws;  ws += (size_t)NN * D * sizeof(float);   // 51.2 MB
  float* hbuf  = (float*)ws;  ws += (size_t)NN * D * sizeof(float);   // 51.2 MB
  float* xpool = (float*)ws;  ws += (size_t)NG * D * sizeof(float);   // 256 KB
  unsigned short* W1b  = (unsigned short*)ws;  ws += (size_t)NL * D * D * 2;
  unsigned short* W2b  = (unsigned short*)ws;  ws += (size_t)NL * D * D * 2;
  unsigned short* PW1b = (unsigned short*)ws;  ws += (size_t)D * D * 2;
  unsigned short* PW2b = (unsigned short*)ws;  ws += (size_t)D * D * 2;

  // convert weights to bf16 (cheap; deterministic every call)
  cvt_bf16_kernel<<<(NL * D * D + 255) / 256, 256, 0, stream>>>(w1, W1b, NL * D * D);
  cvt_bf16_kernel<<<(NL * D * D + 255) / 256, 256, 0, stream>>>(w2, W2b, NL * D * D);
  cvt_bf16_kernel<<<(D * D + 255) / 256, 256, 0, stream>>>(pw1, PW1b, D * D);
  cvt_bf16_kernel<<<(D * D + 255) / 256, 256, 0, stream>>>(pw2, PW2b, D * D);

  const int mlpBlocks = (NN + 127) / 128;   // 782
  const float* hprev = x;
  for (int i = 0; i < NL; ++i) {
    hipMemsetAsync(aggr, 0, (size_t)NN * D * sizeof(float), stream);
    scatter_kernel<<<(NE * 32) / 256, 256, 0, stream>>>(hprev, src, dst, aggr);
    mlp_kernel<<<mlpBlocks, 256, 0, stream>>>(hprev, aggr,
                                              W1b + (size_t)i * D * D, b1 + i * D,
                                              W2b + (size_t)i * D * D, b2 + i * D,
                                              hbuf, NN, /*finalRelu=*/1);
    hprev = hbuf;  // in-place safe: MLP is row-local, staged through LDS
  }

  hipMemsetAsync(xpool, 0, (size_t)NG * D * sizeof(float), stream);
  pool_kernel<<<(NN * 32 + 255) / 256, 256, 0, stream>>>(hbuf, batch, xpool);

  // projection head -> z_out (no final relu), then fc -> p_out
  mlp_kernel<<<NG / 128, 256, 0, stream>>>(xpool, nullptr, PW1b, pb1, PW2b, pb2,
                                           out, NG, /*finalRelu=*/0);
  fc_kernel<<<(NG * NC + 255) / 256, 256, 0, stream>>>(xpool, fw, fbias,
                                                       out + NG * D);
}
